// SBContrastiveLoss_25658134626339
// MI455X (gfx1250) — matrix operand order repaired and checked by reading, not compiled, not run on hardware
//
#include <hip/hip_runtime.h>
#include <hip/hip_bf16.h>

#define NDESC 8192
#define DDIM  128
#define MARGIN_F 2.0f

typedef __bf16 v16bf __attribute__((ext_vector_type(16)));
typedef __bf16 v8bf  __attribute__((ext_vector_type(8)));
typedef float  v8f   __attribute__((ext_vector_type(8)));

// fp32 -> bf16 round-to-nearest-even
__device__ __forceinline__ unsigned short f2bf(float x) {
    unsigned u = __float_as_uint(x);
    u += 0x7FFFu + ((u >> 16) & 1u);
    return (unsigned short)(u >> 16);
}

// order-preserving float -> uint key (so atomicMax on u32 == float max, incl. negatives)
__device__ __forceinline__ unsigned fkey(float x) {
    unsigned u = __float_as_uint(x);
    return (u & 0x80000000u) ? ~u : (u | 0x80000000u);
}
__device__ __forceinline__ float kinv(unsigned k) {
    return (k & 0x80000000u) ? __uint_as_float(k ^ 0x80000000u) : __uint_as_float(~k);
}

__global__ void sb_init_kernel(unsigned* __restrict__ rowkey, unsigned* __restrict__ colkey) {
    int i = blockIdx.x * 256 + threadIdx.x;
    if (i < NDESC) { rowkey[i] = 0u; colkey[i] = 0u; }
}

// One wave32 per row: normalize desc1/desc2 rows to bf16, compute diagonal cos-sim.
__global__ void __launch_bounds__(256)
sb_norm_kernel(const float* __restrict__ d1, const float* __restrict__ d2,
               unsigned short* __restrict__ n1, unsigned short* __restrict__ n2,
               float* __restrict__ same) {
    const int row  = (blockIdx.x * 256 + (int)threadIdx.x) >> 5;
    const int lane = threadIdx.x & 31;

    const float4 a = *(const float4*)(d1 + (size_t)row * DDIM + lane * 4);
    const float4 b = *(const float4*)(d2 + (size_t)row * DDIM + lane * 4);

    float s1 = a.x*a.x + a.y*a.y + a.z*a.z + a.w*a.w;
    float s2 = b.x*b.x + b.y*b.y + b.z*b.z + b.w*b.w;
    float cr = a.x*b.x + a.y*b.y + a.z*b.z + a.w*b.w;
    #pragma unroll
    for (int m = 16; m >= 1; m >>= 1) {
        s1 += __shfl_xor(s1, m, 32);
        s2 += __shfl_xor(s2, m, 32);
        cr += __shfl_xor(cr, m, 32);
    }
    const float inv1 = rsqrtf(s1);
    const float inv2 = rsqrtf(s2);

    uint2 p1, p2;
    p1.x = (unsigned)f2bf(a.x * inv1) | ((unsigned)f2bf(a.y * inv1) << 16);
    p1.y = (unsigned)f2bf(a.z * inv1) | ((unsigned)f2bf(a.w * inv1) << 16);
    p2.x = (unsigned)f2bf(b.x * inv2) | ((unsigned)f2bf(b.y * inv2) << 16);
    p2.y = (unsigned)f2bf(b.z * inv2) | ((unsigned)f2bf(b.w * inv2) << 16);
    *(uint2*)(n1 + (size_t)row * DDIM + lane * 4) = p1;
    *(uint2*)(n2 + (size_t)row * DDIM + lane * 4) = p2;

    if (lane == 0) same[row] = cr * inv1 * inv2;
}

// Fused GEMM + row/col max. Block: 8 waves, tile 128(M) x 64(N).
// Each wave: 16(M) x 64(N) strip = 4 WMMA accumulators, K loop 4 x 32,
// software double-buffered so WMMAs never wait on the latest load clause.
__global__ void __launch_bounds__(256)
sb_gemm_kernel(const unsigned short* __restrict__ n1u, const unsigned short* __restrict__ n2u,
               unsigned* __restrict__ rowkey, unsigned* __restrict__ colkey) {
    const int lane = threadIdx.x & 31;
    const int wave = threadIdx.x >> 5;
    const int h    = lane >> 4;    // half-wave
    const int ln   = lane & 15;
    const int rowBase = blockIdx.y * 128 + wave * 16;
    const int colBase = blockIdx.x * 64;

    const __bf16* A = (const __bf16*)n1u;
    const __bf16* B = (const __bf16*)n2u;

    // Per-lane fragment base pointers (kk advances via +32-element immediate offsets)
    const __bf16* pa = A + (size_t)(rowBase + ln) * DDIM + 16 * h;
    const __bf16* pb[4];
    #pragma unroll
    for (int t = 0; t < 4; ++t)
        pb[t] = B + (size_t)(colBase + 16 * t + ln) * DDIM + 16 * h;

    // Prologue: fragments for kk = 0
    v8bf a_lo = *(const v8bf*)pa;
    v8bf a_hi = *(const v8bf*)(pa + 8);
    v16bf bcur[4];
    #pragma unroll
    for (int t = 0; t < 4; ++t) bcur[t] = *(const v16bf*)pb[t];

    const float NEGINF = -__builtin_inff();
    v8f acc0 = {0.f,0.f,0.f,0.f,0.f,0.f,0.f,0.f};
    v8f acc[4] = {acc0, acc0, acc0, acc0};

    #pragma unroll
    for (int kk = 0; kk < 4; ++kk) {
        v8bf a_lo_n, a_hi_n;
        v16bf bnxt[4];
        if (kk < 3) {   // prefetch kk+1 before consuming kk
            const int kb = (kk + 1) * 32;
            a_lo_n = *(const v8bf*)(pa + kb);
            a_hi_n = *(const v8bf*)(pa + kb + 8);
            #pragma unroll
            for (int t = 0; t < 4; ++t) bnxt[t] = *(const v16bf*)(pb[t] + kb);
        }
        // A fragment (16x32 bf16): lane holds row ln; VGPR0-3: K=16h+0..7, VGPR4-7: K=16h+8..15
        v16bf af = __builtin_shufflevector(a_lo, a_hi,
                       0,1,2,3,4,5,6,7,8,9,10,11,12,13,14,15);
        #pragma unroll
        for (int t = 0; t < 4; ++t)
            acc[t] = __builtin_amdgcn_wmma_f32_16x16x32_bf16(
                         false, af, false, bcur[t], (short)0, acc[t], false, false);
        if (kk < 3) {
            a_lo = a_lo_n; a_hi = a_hi_n;
            #pragma unroll
            for (int t = 0; t < 4; ++t) bcur[t] = bnxt[t];
        }
    }

    // C layout: acc[t][r] is S[rowBase + r + 8h][colBase + 16t + ln]
    float rowAcc[8];
    #pragma unroll
    for (int r = 0; r < 8; ++r) rowAcc[r] = NEGINF;

    #pragma unroll
    for (int t = 0; t < 4; ++t) {
        const int gj = colBase + 16 * t + ln;
        float colPart = NEGINF;
        #pragma unroll
        for (int r = 0; r < 8; ++r) {
            const int gi = rowBase + r + 8 * h;
            float v = acc[t][r];
            if (gi == gj) v = NEGINF;          // exclude diagonal
            rowAcc[r] = fmaxf(rowAcc[r], v);
            colPart   = fmaxf(colPart, v);
        }
        // combine the two half-waves (M=r vs M=r+8) for this column
        colPart = fmaxf(colPart, __shfl_xor(colPart, 16, 32));
        if (h == 0) atomicMax(&colkey[gj], fkey(colPart));
    }

    // reduce each row max across the 16 lanes of the half-wave
    #pragma unroll
    for (int r = 0; r < 8; ++r) {
        float v = rowAcc[r];
        v = fmaxf(v, __shfl_xor(v, 1, 16));
        v = fmaxf(v, __shfl_xor(v, 2, 16));
        v = fmaxf(v, __shfl_xor(v, 4, 16));
        v = fmaxf(v, __shfl_xor(v, 8, 16));
        rowAcc[r] = v;
    }
    if (ln == 0) {
        #pragma unroll
        for (int r = 0; r < 8; ++r)
            atomicMax(&rowkey[rowBase + r + 8 * h], fkey(rowAcc[r]));
    }
}

__global__ void __launch_bounds__(256)
sb_finish_kernel(const float* __restrict__ same, const unsigned* __restrict__ rowkey,
                 const unsigned* __restrict__ colkey, float* __restrict__ out) {
    __shared__ float sm[8];
    float acc = 0.f;
    for (int i = threadIdx.x; i < NDESC; i += 256)
        acc += 2.0f * MARGIN_F - 2.0f * same[i] + kinv(rowkey[i]) + kinv(colkey[i]);
    #pragma unroll
    for (int m = 16; m >= 1; m >>= 1) acc += __shfl_xor(acc, m, 32);
    if ((threadIdx.x & 31) == 0) sm[threadIdx.x >> 5] = acc;
    __syncthreads();
    if (threadIdx.x == 0) {
        float t = 0.f;
        #pragma unroll
        for (int w = 0; w < 8; ++w) t += sm[w];
        out[0] = t * (float)(NDESC - 1) / (float)NDESC;
    }
}

extern "C" void kernel_launch(void* const* d_in, const int* in_sizes, int n_in,
                              void* d_out, int out_size, void* d_ws, size_t ws_size,
                              hipStream_t stream) {
    (void)in_sizes; (void)n_in; (void)out_size; (void)ws_size;
    const float* d1 = (const float*)d_in[0];
    const float* d2 = (const float*)d_in[1];
    float* out = (float*)d_out;

    char* ws = (char*)d_ws;
    const size_t bf16_mat = (size_t)NDESC * DDIM * sizeof(unsigned short); // 2 MB each
    unsigned short* n1   = (unsigned short*)ws;
    unsigned short* n2   = (unsigned short*)(ws + bf16_mat);
    float*          same = (float*)(ws + 2 * bf16_mat);
    unsigned*     rowkey = (unsigned*)(ws + 2 * bf16_mat + NDESC * sizeof(float));
    unsigned*     colkey = rowkey + NDESC;

    sb_init_kernel<<<dim3((NDESC + 255) / 256), dim3(256), 0, stream>>>(rowkey, colkey);
    sb_norm_kernel<<<dim3(NDESC / 8), dim3(256), 0, stream>>>(d1, d2, n1, n2, same);
    sb_gemm_kernel<<<dim3(NDESC / 64, NDESC / 128), dim3(256), 0, stream>>>(n1, n2, rowkey, colkey);
    sb_finish_kernel<<<dim3(1), dim3(256), 0, stream>>>(same, rowkey, colkey, out);
}
